// EdgeTransformer_68925635166338
// MI455X (gfx1250) — matrix operand order repaired and checked by reading, not compiled, not run on hardware
//
#include <hip/hip_runtime.h>
#include <hip/hip_bf16.h>

// ---------------------------------------------------------------------------
// EdgeTransformer (triangle-attention style) for MI455X / gfx1250, wave32.
// All dense contractions go through v_wmma_f32_16x16x32_f16.
// ---------------------------------------------------------------------------

typedef __attribute__((ext_vector_type(16))) _Float16 v16h;
typedef __attribute__((ext_vector_type(8)))  _Float16 v8h;
typedef __attribute__((ext_vector_type(8)))  float    v8f;
typedef __attribute__((ext_vector_type(4)))  float    v4f;

#define Bb   2
#define Nn   192
#define Dd   256
#define NN   (Nn * Nn)          // 36864
#define MTOT (Bb * NN)          // 73728 rows

__device__ inline v8f wmma_f16(v16h a, v16h b, v8f c) {
  return __builtin_amdgcn_wmma_f32_16x16x32_f16(false, a, false, b, (short)0, c,
                                                false, false);
}

// A-matrix fragment (16x32 f16, M rows striped over lanes):
//   lanes 0-15  : K = k0+0..7   (v0..v3), K = k0+16..23 (v4..v7)
//   lanes 16-31 : K = k0+8..15,           K = k0+24..31
__device__ inline v16h frag_ld_a(const _Float16* base, long rowStride,
                                 int k0, int lane) {
  const int r  = lane & 15;
  const int ks = k0 + ((lane & 16) ? 8 : 0);
  const _Float16* p = base + (long)r * rowStride + ks;
  v8h lo = *(const v8h*)(p);
  v8h hi = *(const v8h*)(p + 16);
  v16h f;
#pragma unroll
  for (int t = 0; t < 8; ++t) { f[t] = lo[t]; f[t + 8] = hi[t]; }
  return f;
}

// B-matrix fragment loaded from rows of B^T (32x16 f16, K=32):
//   lanes 0-15  : K = k0+0..7, k0+16..23  (N = lane)
//   lanes 16-31 : K = k0+8..15, k0+24..31 (N = lane-16)
__device__ inline v16h frag_ld_b(const _Float16* __restrict__ baseT, long rowStride,
                                 int k0, int lane) {
  const int r  = lane & 15;
  const int ks = k0 + ((lane & 16) ? 8 : 0);
  const _Float16* p = baseT + (long)r * rowStride + ks;
  v8h lo = *(const v8h*)(p);
  v8h hi = *(const v8h*)(p + 16);
  v16h f;
#pragma unroll
  for (int t = 0; t < 8; ++t) { f[t] = lo[t]; f[t + 8] = hi[t]; }
  return f;
}

// ---------------------------------------------------------------------------
// Kernel 0: convert the five DxD weight matrices to f16 (Bᵀ source for WMMA).
// Slots: 0=Wq 1=Wk 2=Wv 3=Wg 4=Wo
// ---------------------------------------------------------------------------
__global__ void et_prep_weights(const float* __restrict__ wq, const float* __restrict__ wk,
                                const float* __restrict__ wv, const float* __restrict__ wg,
                                const float* __restrict__ wo, _Float16* __restrict__ W16) {
  int idx = blockIdx.x * blockDim.x + threadIdx.x;
  if (idx >= 5 * Dd * Dd) return;
  int s = idx >> 16, o = idx & 65535;
  const float* src = (s == 0) ? wq : (s == 1) ? wk : (s == 2) ? wv : (s == 3) ? wg : wo;
  W16[idx] = (_Float16)src[o];
}

// ---------------------------------------------------------------------------
// Kernel 1: bmat[e] = Edges[e,:].Wb_w + Wb_b   (rank-1 projection, f32 VALU)
// ---------------------------------------------------------------------------
__global__ void et_bmat(const float* __restrict__ E, const float* __restrict__ wb,
                        const float* __restrict__ bb, float* __restrict__ bm) {
  int e = blockIdx.x * blockDim.x + threadIdx.x;
  if (e >= MTOT) return;
  const float* row = E + (long)e * Dd;
  float acc = bb[0];
  for (int k = 0; k < Dd; k += 4) {
    v4f x = *(const v4f*)(row + k);
    v4f w = *(const v4f*)(wb + k);
    acc += x[0] * w[0] + x[1] * w[1] + x[2] * w[2] + x[3] * w[3];
  }
  bm[e] = acc;
}

// ---------------------------------------------------------------------------
// Kernel 2: fused q/k/v/g projection.  One block = one 16-row slab of Edges,
// staged once into LDS as f16 and shared by 8 waves (128-wide output strips
// of the stacked 1024 output cols).  v is written straight into two
// transposed layouts for the output einsums; VT rows pack into 16B stores.
// ---------------------------------------------------------------------------
__global__ void et_proj(const float* __restrict__ E, const _Float16* __restrict__ W16,
                        const float* __restrict__ bq, const float* __restrict__ bk,
                        const float* __restrict__ bv, const float* __restrict__ bg,
                        _Float16* __restrict__ Q16, _Float16* __restrict__ K16,
                        _Float16* __restrict__ VT,  _Float16* __restrict__ VC,
                        _Float16* __restrict__ G16) {
  __shared__ _Float16 As[16 * Dd];            // 8 KB: one 16x256 f16 A slab
  const int tid   = threadIdx.x;
  const int lane  = tid & 31;
  const int wave  = tid >> 5;
  const long row0 = (long)blockIdx.x * 16;
  const int strip = wave * 128;               // global output column base

  // Stage + convert the A slab once (coalesced 64B per thread).
  {
    const float* src = E + row0 * Dd + tid * 16;
    v4f x0 = *(const v4f*)(src);
    v4f x1 = *(const v4f*)(src + 4);
    v4f x2 = *(const v4f*)(src + 8);
    v4f x3 = *(const v4f*)(src + 12);
    v8h h0, h1;
#pragma unroll
    for (int t = 0; t < 4; ++t) {
      h0[t] = (_Float16)x0[t]; h0[t + 4] = (_Float16)x1[t];
      h1[t] = (_Float16)x2[t]; h1[t + 4] = (_Float16)x3[t];
    }
    *(v8h*)(&As[tid * 16])     = h0;
    *(v8h*)(&As[tid * 16 + 8]) = h1;
  }
  __syncthreads();

  v8f acc[8] = {};
  for (int kk = 0; kk < Dd; kk += 32) {
    v16h a = frag_ld_a(As, Dd, kk, lane);
#pragma unroll
    for (int t = 0; t < 8; ++t) {
      int c0  = strip + t * 16;
      int mat = c0 >> 8;                      // 0=q 1=k 2=v 3=g (wave-uniform)
      int lc  = c0 & 255;
      v16h bf = frag_ld_b(W16 + (long)mat * 65536 + (long)lc * Dd, Dd, kk, lane);
      acc[t]  = wmma_f16(a, bf, acc[t]);
    }
  }

  const int hi8 = (lane & 16) ? 8 : 0;
  const int nc  = lane & 15;
  // Decode once: all 16 rows of the slab share (b, i); j0 = first j of my half.
  const long e0 = row0 + hi8;
  const int  bidx = (int)(e0 / NN);
  const int  rem  = (int)(e0 % NN);
  const int  ii   = rem / Nn;
  const int  j0   = rem % Nn;                 // multiple of 8; rows stay in-row
#pragma unroll
  for (int t = 0; t < 8; ++t) {
    int c0  = strip + t * 16;
    int mat = c0 >> 8;
    int cl  = (c0 & 255) + nc;
    float bias = (mat == 0 ? bq : mat == 1 ? bk : mat == 2 ? bv : bg)[cl];
    if (mat == 2) {
      // v: rows r -> consecutive j.  VT contiguous in j: one 16B store.
      v8h pk;
#pragma unroll
      for (int r = 0; r < 8; ++r) pk[r] = (_Float16)(acc[t][r] + bias);
      *(v8h*)&VT[((long)(bidx * Nn + ii) * Dd + cl) * Nn + j0] = pk;
#pragma unroll
      for (int r = 0; r < 8; ++r)             // VC contiguous in i (fixed) -> scatter
        VC[((long)(bidx * Nn + (j0 + r)) * Dd + cl) * Nn + ii] = pk[r];
    } else {
#pragma unroll
      for (int r = 0; r < 8; ++r) {
        long e   = row0 + r + hi8;
        float vv = acc[t][r] + bias;
        if (mat == 0)      Q16[e * Dd + cl] = (_Float16)vv;
        else if (mat == 1) K16[e * Dd + cl] = (_Float16)vv;
        else {
          float s = 1.0f / (1.0f + __expf(-vv));
          G16[e * Dd + cl] = (_Float16)s;
        }
      }
    }
  }
}

// ---------------------------------------------------------------------------
// Kernel 3: row-attention scores  S1 = Q_i K_iᵀ * scale + sym_bias
// One wave per (b, i, j-tile); 12 accumulators cover all 192 l columns.
// ---------------------------------------------------------------------------
__global__ void et_scores1(const _Float16* __restrict__ Q16, const _Float16* __restrict__ K16,
                           const float* __restrict__ BM, float* __restrict__ SC) {
  const int lane = threadIdx.x & 31;
  int wg = blockIdx.x * 8 + (threadIdx.x >> 5);
  int jt = wg % 12, bi = wg / 12;
  int i = bi % Nn, b = bi / Nn;

  const _Float16* Abase = Q16 + ((long)(b * NN + i * Nn + jt * 16)) * Dd;
  v8f acc[12] = {};
  for (int kk = 0; kk < Dd; kk += 32) {
    v16h a = frag_ld_a(Abase, Dd, kk, lane);
#pragma unroll
    for (int lt = 0; lt < 12; ++lt) {
      v16h bt = frag_ld_b(K16 + ((long)(b * NN + i * Nn + lt * 16)) * Dd, Dd, kk, lane);
      acc[lt] = wmma_f16(a, bt, acc[lt]);
    }
  }

  const int hi8 = (lane & 16) ? 8 : 0;
  const int nc  = lane & 15;
  const float scale = 0.0625f;                // 1/sqrt(256)
#pragma unroll
  for (int lt = 0; lt < 12; ++lt) {
    int l = lt * 16 + nc;
    float bias = BM[(long)b * NN + i * Nn + l] + BM[(long)b * NN + (long)l * Nn + i];
#pragma unroll
    for (int r = 0; r < 8; ++r) {
      int j = jt * 16 + r + hi8;
      SC[((long)(b * Nn + i) * Nn + j) * Nn + l] = acc[lt][r] * scale + bias;
    }
  }
}

// ---------------------------------------------------------------------------
// Kernel 4: column-attention scores  S2[i,l] = q[:,j].k[:,j]ᵀ, accumulated
// into SC (exactly one writer per element -> deterministic RMW).
// ---------------------------------------------------------------------------
__global__ void et_scores2(const _Float16* __restrict__ Q16, const _Float16* __restrict__ K16,
                           float* __restrict__ SC) {
  const int lane = threadIdx.x & 31;
  int wg = blockIdx.x * 8 + (threadIdx.x >> 5);
  int it = wg % 12, bj = wg / 12;
  int j = bj % Nn, b = bj / Nn;
  const long rs = (long)Nn * Dd;              // row stride between i (or l) rows

  const _Float16* Abase = Q16 + ((long)(b * NN + it * 16 * Nn + j)) * Dd;
  v8f acc[12] = {};
  for (int kk = 0; kk < Dd; kk += 32) {
    v16h a = frag_ld_a(Abase, rs, kk, lane);
#pragma unroll
    for (int lt = 0; lt < 12; ++lt) {
      v16h bt = frag_ld_b(K16 + ((long)(b * NN + lt * 16 * Nn + j)) * Dd, rs, kk, lane);
      acc[lt] = wmma_f16(a, bt, acc[lt]);
    }
  }

  const int hi8 = (lane & 16) ? 8 : 0;
  const int nc  = lane & 15;
  const float scale = 0.0625f;
#pragma unroll
  for (int lt = 0; lt < 12; ++lt) {
    int l = lt * 16 + nc;
#pragma unroll
    for (int r = 0; r < 8; ++r) {
      int i = it * 16 + r + hi8;
      long idx = ((long)(b * Nn + i) * Nn + j) * Nn + l;
      SC[idx] += acc[lt][r] * scale;
    }
  }
}

// ---------------------------------------------------------------------------
// Kernel 5: softmax over l (192) per (b,i,j) row -> alpha (f16).
// One wave per row, 6 elements per lane, shuffle reductions (wave32).
// ---------------------------------------------------------------------------
__global__ void et_softmax(const float* __restrict__ SC, _Float16* __restrict__ AL) {
  const int lane = threadIdx.x & 31;
  long row = (long)blockIdx.x * 8 + (threadIdx.x >> 5);
  const float* src = SC + row * Nn;
  float v[6];
  float mx = -1e30f;
#pragma unroll
  for (int t = 0; t < 6; ++t) { v[t] = src[lane + t * 32]; mx = fmaxf(mx, v[t]); }
#pragma unroll
  for (int s = 16; s > 0; s >>= 1) mx = fmaxf(mx, __shfl_xor(mx, s, 32));
  float sum = 0.0f;
#pragma unroll
  for (int t = 0; t < 6; ++t) { v[t] = __expf(v[t] - mx); sum += v[t]; }
#pragma unroll
  for (int s = 16; s > 0; s >>= 1) sum += __shfl_xor(sum, s, 32);
  float inv = 1.0f / sum;
#pragma unroll
  for (int t = 0; t < 6; ++t) AL[row * Nn + lane + t * 32] = (_Float16)(v[t] * inv);
}

// ---------------------------------------------------------------------------
// Kernel 6: o1 + o2 fused, gated by sigmoid(g) -> h (f16).
//   o1 = alpha_i @ V_i          (Bᵀ rows from VT)
//   o2 = alpha[:,i,:] @ v[:,i]  (A rows strided, Bᵀ rows from VC)
// One wave per (b, i, j-tile, d-strip of 128); both chains share accumulators.
// ---------------------------------------------------------------------------
__global__ void et_attout(const _Float16* __restrict__ AL, const _Float16* __restrict__ VT,
                          const _Float16* __restrict__ VC, const _Float16* __restrict__ G16,
                          _Float16* __restrict__ H16) {
  const int lane = threadIdx.x & 31;
  int wg = blockIdx.x * 8 + (threadIdx.x >> 5);
  int ds = wg & 1, w2 = wg >> 1;
  int jt = w2 % 12, bi = w2 / 12;
  int i = bi % Nn, b = bi / Nn;

  const _Float16* A1 = AL + ((long)(b * Nn + i) * Nn + jt * 16) * Nn;        // stride Nn
  const _Float16* A2 = AL + ((long)(b * Nn + jt * 16) * Nn + i) * Nn;        // stride Nn*Nn
  const long rsA2 = (long)Nn * Nn;

  v8f acc[8] = {};
  for (int kk = 0; kk < Nn; kk += 32) {
    v16h a1 = frag_ld_a(A1, Nn, kk, lane);
    v16h a2 = frag_ld_a(A2, rsA2, kk, lane);
#pragma unroll
    for (int t = 0; t < 8; ++t) {
      int d0 = ds * 128 + t * 16;
      v16h b1 = frag_ld_b(VT + ((long)(b * Nn + i) * Dd + d0) * Nn, Nn, kk, lane);
      acc[t] = wmma_f16(a1, b1, acc[t]);
      v16h b2 = frag_ld_b(VC + ((long)(b * Nn + i) * Dd + d0) * Nn, Nn, kk, lane);
      acc[t] = wmma_f16(a2, b2, acc[t]);
    }
  }

  const int hi8 = (lane & 16) ? 8 : 0;
  const int nc  = lane & 15;
#pragma unroll
  for (int t = 0; t < 8; ++t) {
    int d = ds * 128 + t * 16 + nc;
#pragma unroll
    for (int r = 0; r < 8; ++r) {
      int j = jt * 16 + r + hi8;
      long e = (long)b * NN + i * Nn + j;
      float gv = (float)G16[e * Dd + d];
      H16[e * Dd + d] = (_Float16)(gv * acc[t][r]);
    }
  }
}

// ---------------------------------------------------------------------------
// Kernel 7: out = relu(h @ Woᵀ + bo), f32 output.
// One block = 16-row slab, 2 waves = two 128-col strips.
// ---------------------------------------------------------------------------
__global__ void et_final(const _Float16* __restrict__ H16, const _Float16* __restrict__ Wo16,
                         const float* __restrict__ bo, float* __restrict__ out) {
  const int lane = threadIdx.x & 31;
  const int ds   = threadIdx.x >> 5;          // 0 or 1
  const long row0 = (long)blockIdx.x * 16;

  v8f acc[8] = {};
  for (int kk = 0; kk < Dd; kk += 32) {
    v16h a = frag_ld_a(H16 + row0 * Dd, Dd, kk, lane);
#pragma unroll
    for (int t = 0; t < 8; ++t) {
      v16h bt = frag_ld_b(Wo16 + (long)(ds * 128 + t * 16) * Dd, Dd, kk, lane);
      acc[t] = wmma_f16(a, bt, acc[t]);
    }
  }

  const int hi8 = (lane & 16) ? 8 : 0;
  const int nc  = lane & 15;
#pragma unroll
  for (int t = 0; t < 8; ++t) {
    int c = ds * 128 + t * 16 + nc;
    float bias = bo[c];
#pragma unroll
    for (int r = 0; r < 8; ++r) {
      long e = row0 + r + hi8;
      float v = acc[t][r] + bias;
      out[e * Dd + c] = v > 0.0f ? v : 0.0f;
    }
  }
}

// ---------------------------------------------------------------------------
extern "C" void kernel_launch(void* const* d_in, const int* in_sizes, int n_in,
                              void* d_out, int out_size, void* d_ws, size_t ws_size,
                              hipStream_t stream) {
  const float* Edges = (const float*)d_in[0];
  const float* Wq_w  = (const float*)d_in[1];
  const float* Wq_b  = (const float*)d_in[2];
  const float* Wk_w  = (const float*)d_in[3];
  const float* Wk_b  = (const float*)d_in[4];
  const float* Wv_w  = (const float*)d_in[5];
  const float* Wv_b  = (const float*)d_in[6];
  const float* Wb_w  = (const float*)d_in[7];
  const float* Wb_b  = (const float*)d_in[8];
  const float* Wg_w  = (const float*)d_in[9];
  const float* Wg_b  = (const float*)d_in[10];
  const float* Wo_w  = (const float*)d_in[11];
  const float* Wo_b  = (const float*)d_in[12];
  float* out = (float*)d_out;

  // Workspace layout (all chunks 256B-aligned; total ~235 MB).
  char* p = (char*)d_ws;
  _Float16* W16 = (_Float16*)p;  p += (size_t)5 * Dd * Dd * 2;   // 655,360
  _Float16* Q16 = (_Float16*)p;  p += (size_t)MTOT * Dd * 2;     // 37.7 MB
  _Float16* K16 = (_Float16*)p;  p += (size_t)MTOT * Dd * 2;
  _Float16* VT  = (_Float16*)p;  p += (size_t)MTOT * Dd * 2;
  _Float16* VC  = (_Float16*)p;  p += (size_t)MTOT * Dd * 2;
  _Float16* G16 = (_Float16*)p;  p += (size_t)MTOT * Dd * 2;
  float*    BM  = (float*)p;     p += (size_t)MTOT * 4;          // 0.3 MB
  float*    SC  = (float*)p;     p += (size_t)Bb * Nn * Nn * Nn * 4; // 56.6 MB
  _Float16* AL  = Q16;   // alpha overlays q (dead after scores)
  _Float16* H16 = K16;   // h overlays k (dead after scores)

  et_prep_weights<<<(5 * Dd * Dd + 255) / 256, 256, 0, stream>>>(
      Wq_w, Wk_w, Wv_w, Wg_w, Wo_w, W16);
  et_bmat<<<(MTOT + 255) / 256, 256, 0, stream>>>(Edges, Wb_w, Wb_b, BM);
  et_proj<<<MTOT / 16, 256, 0, stream>>>(Edges, W16, Wq_b, Wk_b, Wv_b, Wg_b,
                                         Q16, K16, VT, VC, G16);
  et_scores1<<<(Bb * Nn * 12) / 8, 256, 0, stream>>>(Q16, K16, BM, SC);
  et_scores2<<<(Bb * Nn * 12) / 8, 256, 0, stream>>>(Q16, K16, SC);
  et_softmax<<<(Bb * Nn * Nn) / 8, 256, 0, stream>>>(SC, AL);
  et_attout<<<(Bb * Nn * 12 * 2) / 8, 256, 0, stream>>>(AL, VT, VC, G16, H16);
  et_final<<<MTOT / 16, 64, 0, stream>>>(H16, W16 + (size_t)4 * Dd * Dd, Wo_b, out);
}
